// FFM_25451976196515
// MI455X (gfx1250) — matrix operand order repaired and checked by reading, not compiled, not run on hardware
//
#include <hip/hip_runtime.h>

// FFM quadratic form on MI455X (gfx1250, wave32, WMMA bf16)
//   quad[b] = X[b] W X[b]^T,  W = a * (V V^T),  out = sigmoid(quad + X w_lin + b)
// Kernel 1: build W^T in bf16 (padded 845 -> 864) into workspace.
// Kernel 2: fused  X@W  (v_wmma_f32_16x16x32_bf16)  * X  reduction + lin + sigmoid.
//   - 3 j-tiles per K-pass (A-fragment reuse: 2.67 ds_load_b128 per WMMA)
//   - LDS pitch 904 bf16 (452 dwords = 4 mod 64) -> conflict-free b128 banking
//   - W slab staged with global_load_async_to_lds_b128 + s_wait_asynccnt

#define F_   845
#define P_   39
#define B_   32768
#define FP   864              // 27 * 32, padded feature dim
#define LP   904              // LDS row pitch in bf16 (452 dwords == 4 mod 64)
#define NT   (FP / 16)        // 54 j-tiles
#define KT   (FP / 32)        // 27 k-steps
#define NJ   3                // j-tiles per K-pass
#define NO   (NT / NJ)        // 18 outer iterations
#define ROWS 128              // batch rows per workgroup (8 waves x 16)

typedef __attribute__((ext_vector_type(16))) __bf16 v16bf;
typedef __attribute__((ext_vector_type(8)))  float  v8f;

union Frag32 { v16bf v; uint4 q[2]; };   // 32-byte WMMA A/B fragment

__device__ __forceinline__ int field_of(int n) {
    return (n < 13) ? n : ((n - 13) / 32 + 13);
}

// ---------------------------------------------------------------------------
// Kernel 1: WbT[j][i] = bf16( a[i][j] * dot(V[i], V[j]) ), zero-padded to FP.
// ---------------------------------------------------------------------------
__global__ __launch_bounds__(256) void ffm_build_wt(const float* __restrict__ fm,
                                                    const float* __restrict__ a,
                                                    __bf16* __restrict__ WbT) {
    __shared__ float Vi[16][32];
    __shared__ float Vj[16][32];
    const int tid = threadIdx.x;
    const int i0 = blockIdx.x * 16;
    const int j0 = blockIdx.y * 16;

    for (int idx = tid; idx < 16 * 32; idx += 256) {
        int r = idx >> 5, k = idx & 31;
        int gi = i0 + r, gj = j0 + r;
        Vi[r][k] = (gi < F_) ? fm[((size_t)gi * P_ + field_of(gi)) * 32 + k] : 0.0f;
        Vj[r][k] = (gj < F_) ? fm[((size_t)gj * P_ + field_of(gj)) * 32 + k] : 0.0f;
    }
    __syncthreads();

    const int ti = tid & 15;          // i fast -> coalesced WbT row store
    const int tj = tid >> 4;
    const int i = i0 + ti, j = j0 + tj;

    float dot = 0.0f;
#pragma unroll
    for (int k = 0; k < 32; ++k) dot += Vi[ti][k] * Vj[tj][k];

    float w = (i < F_ && j < F_) ? a[(size_t)i * F_ + j] * dot : 0.0f;
    WbT[(size_t)j * FP + i] = (__bf16)w;
}

// ---------------------------------------------------------------------------
// Kernel 2: fused quadratic form.
// LDS: Xh[ROWS][LP] bf16 (231 KB) + Ws[48][LP] bf16 slab (87 KB) = 318 KB.
// ---------------------------------------------------------------------------
__global__ __launch_bounds__(256) void ffm_main(const float* __restrict__ X,
                                                const __bf16* __restrict__ WbT,
                                                const float* __restrict__ w_lin,
                                                const float* __restrict__ b_lin,
                                                float* __restrict__ out) {
    extern __shared__ char smem[];
    __bf16* Xh = (__bf16*)smem;                                        // [ROWS][LP]
    __bf16* Ws = (__bf16*)(smem + (size_t)ROWS * LP * sizeof(__bf16)); // [48][LP]

    const int tid  = threadIdx.x;
    const int wave = tid >> 5;
    const int lane = tid & 31;
    const int half = lane >> 4;       // lane-half selects K sub-blocks / M+8
    const int l15  = lane & 15;
    const int row0 = blockIdx.x * ROWS;

    // ---- stage X tile -> bf16 LDS (zero-pad cols F_..FP-1) ----
    for (int idx = tid; idx < ROWS * FP; idx += 256) {
        int r = idx / FP;
        int c = idx - r * FP;
        float v = (c < F_) ? X[(size_t)(row0 + r) * F_ + c] : 0.0f;
        Xh[r * LP + c] = (__bf16)v;
    }
    __syncthreads();

    // ---- seed accumulators with the linear-term partials ----
    // pacc[v] belongs to batch row  wave*16 + v + half*8 ; summed over c = l15 (mod 16)
    float pacc[8];
#pragma unroll
    for (int v = 0; v < 8; ++v) {
        const int r = wave * 16 + v + half * 8;
        float s = 0.0f;
        for (int c = l15; c < F_; c += 16)
            s += (float)Xh[r * LP + c] * w_lin[c];
        pacc[v] = s;
    }

    const int rowA = wave * 16 + l15;   // A-matrix row owned by this lane

    for (int jo = 0; jo < NO; ++jo) {
        __syncthreads();                // previous slab fully consumed
        // ---- stage 48-row W^T slab via CDNA5 async copy to LDS ----
        {
            const __bf16* src = WbT + (size_t)jo * 48 * FP;
            // 48 rows x 108 uint4 (16 B) each
            for (int idx = tid; idx < 48 * 108; idx += 256) {
                int r = idx / 108, c = idx - r * 108;
                const __bf16* g = src + r * FP + c * 8;
                __bf16*       l = Ws + r * LP + c * 8;
                unsigned laddr = (unsigned)(uintptr_t)l;        // LDS byte address
                unsigned long long gaddr = (unsigned long long)(uintptr_t)g;
                asm volatile("global_load_async_to_lds_b128 %0, %1, off"
                             :: "v"(laddr), "v"(gaddr) : "memory");
            }
            // pull next slab toward L2 while we compute
            if (jo + 1 < NO) {
                const char* nxt = (const char*)(WbT + (size_t)(jo + 1) * 48 * FP);
                __builtin_prefetch(nxt + tid * 324, 0, 1);      // covers 48*FP*2 B
            }
            asm volatile("s_wait_asynccnt 0" ::: "memory");
        }
        __syncthreads();

        // ---- Y[16 x 48] = X_tile @ W[:, jo*48 .. +47]  over 27 K-steps ----
        v8f c0 = {}, c1 = {}, c2 = {};
        for (int kt = 0; kt < KT; ++kt) {
            const int k0 = kt * 32;
            Frag32 af, b0, b1, b2;
            // A (16x32 bf16): half 0 holds K 0..15, half 1 K 16..31
            af.q[0] = *(const uint4*)(&Xh[rowA * LP + k0 + half * 16]);
            af.q[1] = *(const uint4*)(&Xh[rowA * LP + k0 + half * 16 + 8]);
            // B (32x16 bf16): lanes 0-15 hold K 0..15, lanes 16-31 K 16..31; N = l15
            b0.q[0] = *(const uint4*)(&Ws[(l15)      * LP + k0 + half * 16]);
            b0.q[1] = *(const uint4*)(&Ws[(l15)      * LP + k0 + half * 16 + 8]);
            b1.q[0] = *(const uint4*)(&Ws[(l15 + 16) * LP + k0 + half * 16]);
            b1.q[1] = *(const uint4*)(&Ws[(l15 + 16) * LP + k0 + half * 16 + 8]);
            b2.q[0] = *(const uint4*)(&Ws[(l15 + 32) * LP + k0 + half * 16]);
            b2.q[1] = *(const uint4*)(&Ws[(l15 + 32) * LP + k0 + half * 16 + 8]);
            c0 = __builtin_amdgcn_wmma_f32_16x16x32_bf16(
                     false, af.v, false, b0.v, (short)0, c0, false, false);
            c1 = __builtin_amdgcn_wmma_f32_16x16x32_bf16(
                     false, af.v, false, b1.v, (short)0, c1, false, false);
            c2 = __builtin_amdgcn_wmma_f32_16x16x32_bf16(
                     false, af.v, false, b2.v, (short)0, c2, false, false);
        }

        // ---- fuse (X@W) * X : C VGPR v -> row v + half*8, col N = l15 ----
#pragma unroll
        for (int v = 0; v < 8; ++v) {
            const int m = wave * 16 + v + half * 8;
            const int cb = jo * 48 + l15;
            pacc[v] += c0[v] * (float)Xh[m * LP + cb];
            pacc[v] += c1[v] * (float)Xh[m * LP + cb + 16];
            pacc[v] += c2[v] * (float)Xh[m * LP + cb + 32];
        }
    }

    // ---- reduce the 16 N-lanes of each half (wave32 xor tree) ----
#pragma unroll
    for (int off = 1; off < 16; off <<= 1) {
#pragma unroll
        for (int v = 0; v < 8; ++v)
            pacc[v] += __shfl_xor(pacc[v], off, 32);
    }

    if (l15 == 0) {
        const float bb = b_lin[0];
#pragma unroll
        for (int v = 0; v < 8; ++v) {
            const int r = row0 + wave * 16 + v + half * 8;
            const float t = pacc[v] + bb;
            out[r] = 1.0f / (1.0f + __expf(-t));
        }
    }
}

// ---------------------------------------------------------------------------
extern "C" void kernel_launch(void* const* d_in, const int* in_sizes, int n_in,
                              void* d_out, int out_size, void* d_ws, size_t ws_size,
                              hipStream_t stream) {
    const float* X     = (const float*)d_in[0];  // [B, F]
    const float* fm    = (const float*)d_in[1];  // [F, P, K]
    const float* a     = (const float*)d_in[2];  // [F, F]
    const float* w_lin = (const float*)d_in[3];  // [F]
    const float* b_lin = (const float*)d_in[4];  // [1]
    float* out = (float*)d_out;                  // [B]

    __bf16* WbT = (__bf16*)d_ws;                 // [FP, FP] bf16, ~1.42 MB

    ffm_build_wt<<<dim3(FP / 16, FP / 16), 256, 0, stream>>>(fm, a, WbT);

    const size_t smem = (size_t)(ROWS * LP + 48 * LP) * sizeof(__bf16); // 318208 B
    ffm_main<<<B_ / ROWS, 256, smem, stream>>>(X, WbT, w_lin, b_lin, out);
}
